// RecurrentMoE_31181462569501
// MI455X (gfx1250) — compile-verified, matrix operand order checked by reading, not run on hardware
//
#include <hip/hip_runtime.h>
#include <hip/hip_bf16.h>
#include <math.h>

// Problem constants (match reference)
#define BB 4
#define TT 1024
#define DD 1024
#define HH 16
#define SS 4
#define LK (SS + TT)   // 1028
#define HD (DD / HH)   // 64

typedef _Float16 v8h  __attribute__((ext_vector_type(8)));
typedef _Float16 v16h __attribute__((ext_vector_type(16)));
typedef float    v8f  __attribute__((ext_vector_type(8)));
typedef unsigned u32x4 __attribute__((ext_vector_type(4)));
typedef unsigned u32x8 __attribute__((ext_vector_type(8)));

// ---------------- fourier positional encoding ----------------
__global__ void k_pe(float* __restrict__ pe) {
  int idx = blockIdx.x * blockDim.x + threadIdx.x;
  const int half = DD / 2;
  if (idx >= TT * half) return;
  int t = idx / half;
  int i = idx % half;
  float freq = powf((float)SS, (float)i / (float)half);
  float ang = (float)t / freq;
  pe[(size_t)t * DD + i] = sinf(ang);
  pe[(size_t)t * DD + half + i] = cosf(ang);
}

// ---------------- f32 -> f16 convert ----------------
__global__ void k_cvt(const float* __restrict__ in, _Float16* __restrict__ out, int n) {
  int idx = blockIdx.x * blockDim.x + threadIdx.x;
  if (idx < n) out[idx] = (_Float16)in[idx];
}

// ---------------- TDM weight prefetch: DMA one 128x32 f32 tile into LDS ----------------
// Warms L2 with weight bytes ahead of the f16 conversion + GEMM passes using the
// Tensor Data Mover (TENSORcnt-tracked). Descriptor per CDNA5 D# group0/group1.
#define BM 128
#define BN 128
#define BKK 32
#define LSTR 48   // halves per LDS row (96B, 16B-aligned chunks)

__global__ __launch_bounds__(32) void k_tdm_prefetch(const float* __restrict__ W) {
  __shared__ float stage[BM * BKK];   // 128 x 32 f32 tile (16 KB)
  unsigned long long ga = (unsigned long long)W;
  unsigned lds = (unsigned)(size_t)&stage[0];
  u32x4 g0;
  g0[0] = 1u;                                   // count=1 (valid), user mode
  g0[1] = lds;                                  // lds_addr
  g0[2] = (unsigned)(ga & 0xFFFFFFFFu);         // global_addr[31:0]
  g0[3] = (unsigned)((ga >> 32) & 0x1FFFFFFu)   // global_addr[56:32]
        | (2u << 30);                           // type = 2 ("image")
  u32x8 g1;
  g1[0] = (2u << 16);                                       // data_size=2 -> 4B, wg_mask=0
  g1[1] = ((unsigned)DD & 0xFFFFu) << 16;                   // tensor_dim0[15:0]
  g1[2] = (((unsigned)DD >> 16) & 0xFFFFu)                  // tensor_dim0[31:16]
        | (((unsigned)DD & 0xFFFFu) << 16);                 // tensor_dim1[15:0]
  g1[3] = (((unsigned)DD >> 16) & 0xFFFFu)                  // tensor_dim1[31:16]
        | (32u << 16);                                      // tile_dim0 = 32
  g1[4] = 128u;                                             // tile_dim1 = 128, tile_dim2 = 0
  g1[5] = (unsigned)DD;                                     // tensor_dim0_stride[31:0]
  g1[6] = 0u;                                               // stride hi | dim1_stride lo
  g1[7] = 0u;                                               // dim1_stride hi
  asm volatile("tensor_load_to_lds %0, %1" :: "s"(g0), "s"(g1) : "memory");
  __builtin_amdgcn_s_wait_tensorcnt(0);
}

// ---------------- WMMA tiled GEMM: C[M,N] = A[M,K] @ W[N,K]^T (+bias +pe) ----------------
// Double-buffered LDS with GLOBAL_LOAD_ASYNC_TO_LDS_B128 (ASYNCcnt): DMA of the
// next K-tile overlaps the current tile's v_wmma_f32_16x16x32_f16 work.
__global__ __launch_bounds__(256) void k_gemm(
    const _Float16* __restrict__ A, const _Float16* __restrict__ W,
    const float* __restrict__ bias, const float* __restrict__ pe,
    float* __restrict__ C, int M, int N, int K, int Tmod)
{
  __shared__ _Float16 As[2][BM * LSTR];
  __shared__ _Float16 Bs[2][BN * LSTR];
  const int tid = threadIdx.x;
  const int m0 = blockIdx.x * BM;
  const int n0 = blockIdx.y * BN;
  const int wave = tid >> 5;          // 8 waves (wave32)
  const int lane = tid & 31;
  const int wm = (wave & 3) * 32;     // 4 waves along M
  const int wn = (wave >> 2) * 64;    // 2 waves along N
  const int l16 = lane & 15;
  const int hf  = lane >> 4;

  const v8f vzero = {0.f,0.f,0.f,0.f,0.f,0.f,0.f,0.f};
  v8f acc[2][4];
  for (int i = 0; i < 2; ++i)
    for (int j = 0; j < 4; ++j)
      acc[i][j] = vzero;

  // cooperative tile loader: each thread DMAs 32B of A and 32B of W per K-step
  const int lrow = tid >> 1;          // 0..127
  const int lcol = (tid & 1) * 16;    // 0 or 16 halves
  const int gm = m0 + lrow;
  const int gn = n0 + lrow;
  const uint4 uz = make_uint4(0u, 0u, 0u, 0u);

  auto issue = [&](int k0, int buf) {
    unsigned la = (unsigned)(size_t)&As[buf][lrow * LSTR + lcol];
    unsigned lb = (unsigned)(size_t)&Bs[buf][lrow * LSTR + lcol];
    if (gm < M) {
      unsigned long long ga = (unsigned long long)(A + (size_t)gm * K + k0 + lcol);
      asm volatile(
          "global_load_async_to_lds_b128 %0, %1, off\n\t"
          "global_load_async_to_lds_b128 %0, %1, off offset:16"
          :: "v"(la), "v"(ga) : "memory");
    } else {  // zero-fill out-of-range rows so WMMA sees zeros
      *(uint4*)(&As[buf][lrow * LSTR + lcol])     = uz;
      *(uint4*)(&As[buf][lrow * LSTR + lcol + 8]) = uz;
    }
    if (gn < N) {
      unsigned long long gb = (unsigned long long)(W + (size_t)gn * K + k0 + lcol);
      asm volatile(
          "global_load_async_to_lds_b128 %0, %1, off\n\t"
          "global_load_async_to_lds_b128 %0, %1, off offset:16"
          :: "v"(lb), "v"(gb) : "memory");
    } else {
      *(uint4*)(&Bs[buf][lrow * LSTR + lcol])     = uz;
      *(uint4*)(&Bs[buf][lrow * LSTR + lcol + 8]) = uz;
    }
  };

  issue(0, 0);                         // prologue: DMA tile 0
  const int nk = K / BKK;
  for (int kt = 0; kt < nk; ++kt) {
    const int buf = kt & 1;
    asm volatile("s_wait_asynccnt 0x0" ::: "memory");  // this wave's DMA landed
    __syncthreads();                                   // everyone's landed / reads drained
    if (kt + 1 < nk) issue((kt + 1) * BKK, buf ^ 1);   // overlap next DMA with compute

    // Per-spec 16-bit A/B fragment layout:
    // lanes 0-15 hold K {0..7, 16..23}, lanes 16-31 hold K {8..15, 24..31}
    v16h af[2], bf[4];
    for (int i = 0; i < 2; ++i) {
      const int row = wm + i * 16 + l16;
      v8h lo8 = *(const v8h*)(&As[buf][row * LSTR + hf * 8]);
      v8h hi8 = *(const v8h*)(&As[buf][row * LSTR + hf * 8 + 16]);
      v16h f;
      for (int j = 0; j < 8; ++j) { f[j] = lo8[j]; f[j + 8] = hi8[j]; }
      af[i] = f;
    }
    for (int i = 0; i < 4; ++i) {
      const int row = wn + i * 16 + l16;
      v8h lo8 = *(const v8h*)(&Bs[buf][row * LSTR + hf * 8]);
      v8h hi8 = *(const v8h*)(&Bs[buf][row * LSTR + hf * 8 + 16]);
      v16h f;
      for (int j = 0; j < 8; ++j) { f[j] = lo8[j]; f[j + 8] = hi8[j]; }
      bf[i] = f;
    }
    for (int i = 0; i < 2; ++i)
      for (int j = 0; j < 4; ++j)
        acc[i][j] = __builtin_amdgcn_wmma_f32_16x16x32_f16(
            false, af[i], false, bf[j], (short)0, acc[i][j], false, false);
  }

  // C/D layout: VGPR r -> M = r (+8 for lanes 16-31), N = lane&15
  for (int i = 0; i < 2; ++i) {
    for (int j = 0; j < 4; ++j) {
      const int n = n0 + wn + j * 16 + l16;
      for (int r = 0; r < 8; ++r) {
        const int m = m0 + wm + i * 16 + hf * 8 + r;
        if (m < M && n < N) {
          float v = acc[i][j][r];
          if (bias) v += bias[n];
          if (Tmod) v += pe[(size_t)(m % Tmod) * N + n];
          C[(size_t)m * N + n] = v;
        }
      }
    }
  }
}

// ---------------- block reduction (blockDim == 256) ----------------
__device__ __forceinline__ float blk_sum(float v, float* red) {
  const int tid = threadIdx.x;
  red[tid] = v; __syncthreads();
  for (int off = 128; off > 0; off >>= 1) {
    if (tid < off) red[tid] += red[tid + off];
    __syncthreads();
  }
  float r = red[0]; __syncthreads();
  return r;
}

// ---------------- layernorm over D=1024 for concat([first, second]) rows ----------------
__global__ __launch_bounds__(256) void k_ln(
    const float* __restrict__ first, int n_first, long stride_first,
    const float* __restrict__ second, int n_second, long stride_second,
    const float* __restrict__ g, const float* __restrict__ be,
    float* __restrict__ out_f32, _Float16* __restrict__ out_f16)
{
  __shared__ float red[256];
  const int row = blockIdx.x;
  const int rpb = n_first + n_second;
  const int b = row / rpb;
  const int r = row % rpb;
  const float* src = (r < n_first)
      ? (first + (size_t)b * stride_first + (size_t)r * DD)
      : (second + (size_t)b * stride_second + (size_t)(r - n_first) * DD);
  const int tid = threadIdx.x;
  float x[4];
  float s = 0.f;
  for (int i = 0; i < 4; ++i) { x[i] = src[tid + i * 256]; s += x[i]; }
  const float mu = blk_sum(s, red) * (1.f / DD);
  float v = 0.f;
  for (int i = 0; i < 4; ++i) { float d = x[i] - mu; v += d * d; }
  const float rstd = rsqrtf(blk_sum(v, red) * (1.f / DD) + 1e-5f);
  for (int i = 0; i < 4; ++i) {
    const int c = tid + i * 256;
    const float y = (x[i] - mu) * rstd * g[c] + be[c];
    if (out_f32) out_f32[(size_t)row * DD + c] = y;
    if (out_f16) out_f16[(size_t)row * DD + c] = (_Float16)y;
  }
}

// ---------------- small fp32 GEMM (M<=16): C = A @ W^T + bias [gelu] [+res] ----------------
__global__ __launch_bounds__(256) void k_smallmm(
    const float* __restrict__ A, const float* __restrict__ W,
    const float* __restrict__ bias,
    const float* __restrict__ res, long res_stride,
    float* __restrict__ C, int M, int N, int K, int act)
{
  const int idx = blockIdx.x * blockDim.x + threadIdx.x;
  if (idx >= M * N) return;
  const int m = idx / N, n = idx % N;
  const float* a = A + (size_t)m * K;
  const float* w = W + (size_t)n * K;
  float s = 0.f;
  for (int k = 0; k < K; ++k) s = fmaf(a[k], w[k], s);
  if (bias) s += bias[n];
  if (act == 1) s = 0.5f * s * (1.f + erff(s * 0.7071067811865476f));
  if (res) s += res[(size_t)m * res_stride + n];
  C[idx] = s;
}

// ---------------- small-Lq attention: one block per (b,h,qi), Lk=1028 ----------------
__global__ __launch_bounds__(128) void k_attn(
    const float* __restrict__ Q, const float* __restrict__ Kh,
    const float* __restrict__ Vh, float* __restrict__ O,
    int Lq, int Lk)
{
  __shared__ float qs[HD];
  __shared__ float sc[LK];
  __shared__ float red[128];
  const int blk = blockIdx.x;
  const int qi = blk % Lq;
  const int h  = (blk / Lq) % HH;
  const int b  = blk / (Lq * HH);
  const int tid = threadIdx.x;
  const float* qrow = Q + (size_t)(b * Lq + qi) * DD + h * HD;
  if (tid < HD) qs[tid] = qrow[tid];
  __syncthreads();
  const float scale = rsqrtf((float)HD);
  float lmax = -1e30f;
  for (int k = tid; k < Lk; k += 128) {
    const float* kr = Kh + (size_t)(b * Lk + k) * DD + h * HD;
    float s = 0.f;
    for (int d = 0; d < HD; ++d) s = fmaf(qs[d], kr[d], s);
    s *= scale;
    sc[k] = s;
    lmax = fmaxf(lmax, s);
  }
  red[tid] = lmax; __syncthreads();
  for (int off = 64; off > 0; off >>= 1) {
    if (tid < off) red[tid] = fmaxf(red[tid], red[tid + off]);
    __syncthreads();
  }
  const float mx = red[0]; __syncthreads();
  float lsum = 0.f;
  for (int k = tid; k < Lk; k += 128) {
    const float p = expf(sc[k] - mx);
    sc[k] = p;
    lsum += p;
  }
  red[tid] = lsum; __syncthreads();
  for (int off = 64; off > 0; off >>= 1) {
    if (tid < off) red[tid] += red[tid + off];
    __syncthreads();
  }
  const float denom = red[0]; __syncthreads();
  for (int d = tid; d < HD; d += 128) {
    float s = 0.f;
    for (int k = 0; k < Lk; ++k)
      s = fmaf(sc[k], Vh[(size_t)(b * Lk + k) * DD + h * HD + d], s);
    O[(size_t)(b * Lq + qi) * DD + h * HD + d] = s / denom;
  }
}

// ---------------- top-2 slot gate + state update (one block per batch) ----------------
__global__ __launch_bounds__(256) void k_gate(
    const float* __restrict__ latent, const float* __restrict__ state_old,
    const float* __restrict__ g, const float* __restrict__ be,
    const float* __restrict__ ctxW,
    float* __restrict__ state_ws, float* __restrict__ out)
{
  __shared__ float red[256];
  __shared__ float logits[SS];
  __shared__ float alpha[SS];
  const int b = blockIdx.x;
  const int tid = threadIdx.x;
  for (int s = 0; s < SS; ++s) {
    const float* row = latent + (size_t)(b * SS + s) * DD;
    float x[4]; float sum = 0.f;
    for (int i = 0; i < 4; ++i) { x[i] = row[tid + i * 256]; sum += x[i]; }
    const float mu = blk_sum(sum, red) * (1.f / DD);
    float v = 0.f;
    for (int i = 0; i < 4; ++i) { float d = x[i] - mu; v += d * d; }
    const float rstd = rsqrtf(blk_sum(v, red) * (1.f / DD) + 1e-5f);
    float dot = 0.f;
    for (int i = 0; i < 4; ++i) {
      const int c = tid + i * 256;
      dot += ((x[i] - mu) * rstd * g[c] + be[c]) * ctxW[c];
    }
    const float lg = blk_sum(dot, red);
    if (tid == 0) logits[s] = lg;
    __syncthreads();
  }
  if (tid == 0) {
    int i0 = 0;
    for (int s2 = 1; s2 < SS; ++s2) if (logits[s2] > logits[i0]) i0 = s2;
    int i1 = (i0 == 0) ? 1 : 0;
    for (int s2 = 0; s2 < SS; ++s2)
      if (s2 != i0 && logits[s2] > logits[i1]) i1 = s2;
    const float v0 = logits[i0], v1 = logits[i1];
    const float e0 = 1.f, e1 = expf(v1 - v0);
    const float sm = e0 + e1;
    for (int s2 = 0; s2 < SS; ++s2) alpha[s2] = 0.f;
    alpha[i0] = e0 / sm;
    alpha[i1] += e1 / sm;
    int* iout = (int*)out + (BB * DD) + b * 2;   // tgt_idx region
    iout[0] = i0; iout[1] = i1;
  }
  __syncthreads();
  for (int idx = tid; idx < SS * DD; idx += 256) {
    const int s = idx / DD;
    const float nv = 0.6f * state_old[(size_t)b * SS * DD + idx]
        + 0.4f * alpha[s] * tanhf(latent[(size_t)b * SS * DD + idx]);
    state_ws[(size_t)b * SS * DD + idx] = nv;
    out[(size_t)(BB * DD) + BB * 2 + (size_t)b * SS * DD + idx] = nv;  // state region
  }
}

extern "C" void kernel_launch(void* const* d_in, const int* in_sizes, int n_in,
                              void* d_out, int out_size, void* d_ws, size_t ws_size,
                              hipStream_t stream) {
  (void)in_sizes; (void)n_in; (void)out_size; (void)ws_size;
  const float* x          = (const float*)d_in[0];
  const float* state0     = (const float*)d_in[1];
  const float* se_W       = (const float*)d_in[2];
  const float* se_b       = (const float*)d_in[3];
  const float* s_Wq       = (const float*)d_in[4];
  const float* s_bq       = (const float*)d_in[5];
  const float* s_Wk       = (const float*)d_in[6];
  const float* s_bk       = (const float*)d_in[7];
  const float* s_Wv       = (const float*)d_in[8];
  const float* s_bv       = (const float*)d_in[9];
  const float* s_Wo       = (const float*)d_in[10];
  const float* s_bo       = (const float*)d_in[11];
  const float* s_lnq_g    = (const float*)d_in[12];
  const float* s_lnq_b    = (const float*)d_in[13];
  const float* s_lnkv_g   = (const float*)d_in[14];
  const float* s_lnkv_b   = (const float*)d_in[15];
  const float* s_fc1_W    = (const float*)d_in[16];
  const float* s_fc1_b    = (const float*)d_in[17];
  const float* s_fc2_W    = (const float*)d_in[18];
  const float* s_fc2_b    = (const float*)d_in[19];
  const float* s_lnffn_g  = (const float*)d_in[20];
  const float* s_lnffn_b  = (const float*)d_in[21];
  const float* s_lnslot_g = (const float*)d_in[22];
  const float* s_lnslot_b = (const float*)d_in[23];
  const float* slot_ctx_W = (const float*)d_in[24];
  const float* oe_W       = (const float*)d_in[25];
  const float* oe_b       = (const float*)d_in[26];
  const float* o_Wq       = (const float*)d_in[27];
  const float* o_bq       = (const float*)d_in[28];
  const float* o_Wk       = (const float*)d_in[29];
  const float* o_bk       = (const float*)d_in[30];
  const float* o_Wv       = (const float*)d_in[31];
  const float* o_bv       = (const float*)d_in[32];
  const float* o_Wo       = (const float*)d_in[33];
  const float* o_bo       = (const float*)d_in[34];
  const float* o_lnq_g    = (const float*)d_in[35];
  const float* o_lnq_b    = (const float*)d_in[36];
  const float* o_lnkv_g   = (const float*)d_in[37];
  const float* o_lnkv_b   = (const float*)d_in[38];
  const float* o_fc1_W    = (const float*)d_in[39];
  const float* o_fc1_b    = (const float*)d_in[40];
  const float* o_fc2_W    = (const float*)d_in[41];
  const float* o_fc2_b    = (const float*)d_in[42];
  const float* o_lnffn_g  = (const float*)d_in[43];
  const float* o_lnffn_b  = (const float*)d_in[44];
  const float* op_W       = (const float*)d_in[45];
  const float* op_b       = (const float*)d_in[46];
  float* yout = (float*)d_out;

  char* ws = (char*)d_ws;
  size_t off = 0;
  auto alloc = [&](size_t bytes) -> char* {
    char* p = ws + off;
    off += (bytes + 255) & ~(size_t)255;
    return p;
  };
  float*    pe_f  = (float*)   alloc((size_t)TT * DD * 4);
  _Float16* x_h   = (_Float16*)alloc((size_t)BB * TT * DD * 2);
  _Float16* seW_h = (_Float16*)alloc((size_t)DD * DD * 2);
  _Float16* sWk_h = (_Float16*)alloc((size_t)DD * DD * 2);
  _Float16* sWv_h = (_Float16*)alloc((size_t)DD * DD * 2);
  _Float16* oeW_h = (_Float16*)alloc((size_t)DD * DD * 2);
  _Float16* oWk_h = (_Float16*)alloc((size_t)DD * DD * 2);
  _Float16* oWv_h = (_Float16*)alloc((size_t)DD * DD * 2);
  float*    lx    = (float*)   alloc((size_t)BB * TT * DD * 4);
  _Float16* kvs_h = (_Float16*)alloc((size_t)BB * LK * DD * 2);
  float*    khs   = (float*)   alloc((size_t)BB * LK * DD * 4);
  float*    vhs   = (float*)   alloc((size_t)BB * LK * DD * 4);
  float*    qlns  = (float*)   alloc((size_t)BB * SS * DD * 4);
  float*    qs_   = (float*)   alloc((size_t)BB * SS * DD * 4);
  float*    sattn = (float*)   alloc((size_t)BB * SS * DD * 4);
  float*    lat1  = (float*)   alloc((size_t)BB * SS * DD * 4);
  float*    lat1l = (float*)   alloc((size_t)BB * SS * DD * 4);
  float*    t2    = (float*)   alloc((size_t)BB * SS * DD * 4);
  float*    latent= (float*)   alloc((size_t)BB * SS * DD * 4);
  float*    st_ws = (float*)   alloc((size_t)BB * SS * DD * 4);
  float*    lo    = (float*)   alloc((size_t)BB * TT * DD * 4);
  _Float16* kv2_h = (_Float16*)alloc((size_t)BB * LK * DD * 2);
  float*    kh2   = (float*)   alloc((size_t)BB * LK * DD * 4);
  float*    vh2   = (float*)   alloc((size_t)BB * LK * DD * 4);
  float*    loln  = (float*)   alloc((size_t)BB * DD * 4);
  float*    q2    = (float*)   alloc((size_t)BB * DD * 4);
  float*    attn2 = (float*)   alloc((size_t)BB * DD * 4);
  float*    o1    = (float*)   alloc((size_t)BB * DD * 4);
  float*    o1ln  = (float*)   alloc((size_t)BB * DD * 4);
  float*    t3    = (float*)   alloc((size_t)BB * DD * 4);
  float*    o2    = (float*)   alloc((size_t)BB * DD * 4);

  // 0) TDM prefetch of the six big-GEMM weights (warms L2 via tensor DMA)
  k_tdm_prefetch<<<dim3(1), dim3(32), 0, stream>>>(se_W);
  k_tdm_prefetch<<<dim3(1), dim3(32), 0, stream>>>(s_Wk);
  k_tdm_prefetch<<<dim3(1), dim3(32), 0, stream>>>(s_Wv);
  k_tdm_prefetch<<<dim3(1), dim3(32), 0, stream>>>(oe_W);
  k_tdm_prefetch<<<dim3(1), dim3(32), 0, stream>>>(o_Wk);
  k_tdm_prefetch<<<dim3(1), dim3(32), 0, stream>>>(o_Wv);

  // 1) positional encoding
  k_pe<<<dim3((TT * (DD / 2) + 255) / 256), dim3(256), 0, stream>>>(pe_f);
  // 2) f16 conversions for WMMA GEMM operands
  const int nx = BB * TT * DD, nw = DD * DD;
  k_cvt<<<dim3((nx + 255) / 256), dim3(256), 0, stream>>>(x, x_h, nx);
  k_cvt<<<dim3((nw + 255) / 256), dim3(256), 0, stream>>>(se_W, seW_h, nw);
  k_cvt<<<dim3((nw + 255) / 256), dim3(256), 0, stream>>>(s_Wk, sWk_h, nw);
  k_cvt<<<dim3((nw + 255) / 256), dim3(256), 0, stream>>>(s_Wv, sWv_h, nw);
  k_cvt<<<dim3((nw + 255) / 256), dim3(256), 0, stream>>>(oe_W, oeW_h, nw);
  k_cvt<<<dim3((nw + 255) / 256), dim3(256), 0, stream>>>(o_Wk, oWk_h, nw);
  k_cvt<<<dim3((nw + 255) / 256), dim3(256), 0, stream>>>(o_Wv, oWv_h, nw);

  const dim3 blk256(256);
  const dim3 gTok((BB * TT + BM - 1) / BM, DD / BN);  // 32 x 8
  const dim3 gKV ((BB * LK + BM - 1) / BM, DD / BN);  // 33 x 8

  // 3) lx = x @ se_W^T + se_b + pe
  k_gemm<<<gTok, blk256, 0, stream>>>(x_h, seW_h, se_b, pe_f, lx, BB * TT, DD, DD, TT);

  // ---- state branch ----
  k_ln<<<dim3(BB * SS), blk256, 0, stream>>>(state0, SS, (long)SS * DD, nullptr, 0, 0,
                                             s_lnq_g, s_lnq_b, qlns, nullptr);
  k_smallmm<<<dim3((BB * SS * DD + 255) / 256), blk256, 0, stream>>>(
      qlns, s_Wq, s_bq, nullptr, 0, qs_, BB * SS, DD, DD, 0);
  k_ln<<<dim3(BB * LK), blk256, 0, stream>>>(state0, SS, (long)SS * DD, lx, TT, (long)TT * DD,
                                             s_lnkv_g, s_lnkv_b, nullptr, kvs_h);
  k_gemm<<<gKV, blk256, 0, stream>>>(kvs_h, sWk_h, s_bk, nullptr, khs, BB * LK, DD, DD, 0);
  k_gemm<<<gKV, blk256, 0, stream>>>(kvs_h, sWv_h, s_bv, nullptr, vhs, BB * LK, DD, DD, 0);
  k_attn<<<dim3(BB * HH * SS), dim3(128), 0, stream>>>(qs_, khs, vhs, sattn, SS, LK);
  k_smallmm<<<dim3((BB * SS * DD + 255) / 256), blk256, 0, stream>>>(
      sattn, s_Wo, s_bo, state0, (long)DD, lat1, BB * SS, DD, DD, 0);
  k_ln<<<dim3(BB * SS), blk256, 0, stream>>>(lat1, SS, (long)SS * DD, nullptr, 0, 0,
                                             s_lnffn_g, s_lnffn_b, lat1l, nullptr);
  k_smallmm<<<dim3((BB * SS * DD + 255) / 256), blk256, 0, stream>>>(
      lat1l, s_fc1_W, s_fc1_b, nullptr, 0, t2, BB * SS, DD, DD, 1);
  k_smallmm<<<dim3((BB * SS * DD + 255) / 256), blk256, 0, stream>>>(
      t2, s_fc2_W, s_fc2_b, lat1, (long)DD, latent, BB * SS, DD, DD, 0);
  k_gate<<<dim3(BB), blk256, 0, stream>>>(latent, state0, s_lnslot_g, s_lnslot_b,
                                          slot_ctx_W, st_ws, yout);

  // ---- out branch (only last token feeds y) ----
  k_gemm<<<gTok, blk256, 0, stream>>>(x_h, oeW_h, oe_b, pe_f, lo, BB * TT, DD, DD, TT);
  k_ln<<<dim3(BB * LK), blk256, 0, stream>>>(lo, TT, (long)TT * DD, st_ws, SS, (long)SS * DD,
                                             o_lnkv_g, o_lnkv_b, nullptr, kv2_h);
  k_gemm<<<gKV, blk256, 0, stream>>>(kv2_h, oWk_h, o_bk, nullptr, kh2, BB * LK, DD, DD, 0);
  k_gemm<<<gKV, blk256, 0, stream>>>(kv2_h, oWv_h, o_bv, nullptr, vh2, BB * LK, DD, DD, 0);
  k_ln<<<dim3(BB), blk256, 0, stream>>>(lo + (size_t)(TT - 1) * DD, 1, (long)TT * DD,
                                        nullptr, 0, 0, o_lnq_g, o_lnq_b, loln, nullptr);
  k_smallmm<<<dim3((BB * DD + 255) / 256), blk256, 0, stream>>>(
      loln, o_Wq, o_bq, nullptr, 0, q2, BB, DD, DD, 0);
  k_attn<<<dim3(BB * HH), dim3(128), 0, stream>>>(q2, kh2, vh2, attn2, 1, LK);
  k_smallmm<<<dim3((BB * DD + 255) / 256), blk256, 0, stream>>>(
      attn2, o_Wo, o_bo, lo + (size_t)(TT - 1) * DD, (long)TT * DD, o1, BB, DD, DD, 0);
  k_ln<<<dim3(BB), blk256, 0, stream>>>(o1, 1, (long)DD, nullptr, 0, 0,
                                        o_lnffn_g, o_lnffn_b, o1ln, nullptr);
  k_smallmm<<<dim3((BB * DD + 255) / 256), blk256, 0, stream>>>(
      o1ln, o_fc1_W, o_fc1_b, nullptr, 0, t3, BB, DD, DD, 1);
  k_smallmm<<<dim3((BB * DD + 255) / 256), blk256, 0, stream>>>(
      t3, o_fc2_W, o_fc2_b, o1, (long)DD, o2, BB, DD, DD, 0);
  // y -> d_out[0 .. B*D)
  k_smallmm<<<dim3((BB * DD + 255) / 256), blk256, 0, stream>>>(
      o2, op_W, op_b, nullptr, 0, yout, BB, DD, DD, 0);
}
